// BiUpsampler_54829552500776
// MI455X (gfx1250) — matrix operand order repaired and checked by reading, not compiled, not run on hardware
//
#include <hip/hip_runtime.h>
#include <hip/hip_bf16.h>

typedef __attribute__((ext_vector_type(16))) __bf16 v16bf;
typedef __attribute__((ext_vector_type(8)))  __bf16 v8bf;
typedef __attribute__((ext_vector_type(8)))  float  v8f;

#define C_CH   64
#define HI_IN  256
#define WI_IN  256
#define HWIN   (HI_IN * WI_IN)
#define HOWO   (512 * 512)
#define NPIX   64          // output pixels per block
#define NBLKS  (HOWO / NPIX)

// ---- LDS layout (bytes) ----
#define ACT1_OFF 0          // 64 * 64  bf16 =  8192
#define ACT2_OFF 8192       // 64 * 256 bf16 = 32768
#define DWB_OFF  40960      // 64 * 576 bf16 = 73728
#define PWB_OFF  114688     // 64 * 192 bf16 = 24576
#define OUTB_OFF 139264     // 64 * 3   f32  =   768
#define SMEM_BYTES 140032

// ---- bf16 weight workspace layout (elements) ----
#define WS_DW2 0            // 256*64
#define WS_PW2 16384        // 256*64
#define WS_DW3 32768        // 576*256
#define WS_PW3 180224       // 192*256
#define WS_ELEMS 229376

// Convert fp32 weights -> bf16 into workspace (runs each launch; deterministic).
__global__ void biup_cvt_weights(const float* __restrict__ dw2,
                                 const float* __restrict__ dw3,
                                 const float* __restrict__ pw2,
                                 const float* __restrict__ pw3,
                                 __bf16* __restrict__ ws) {
  int i = blockIdx.x * 256 + threadIdx.x;
  if (i < 16384)            ws[i] = (__bf16)dw2[i];
  else if (i < 32768)       ws[i] = (__bf16)pw2[i - 16384];
  else if (i < 180224)      ws[i] = (__bf16)dw3[i - 32768];
  else if (i < WS_ELEMS)    ws[i] = (__bf16)pw3[i - 180224];
}

// One wave computes MT 16x16 output tiles (fixed pixel tile, M-range m0..m0+16*MT)
// A: global bf16 weights, row-major [O][KD].
//    16-bit A layout: lane l<16 row M=l holds K {kk..kk+7, kk+16..kk+23},
//                     lane l>=16 same row,  K {kk+8..kk+15, kk+24..kk+31}.
// B: LDS bf16 activations [pixel][KD]; lane = column(pixel), same K layout as A
//    (lo lanes K {kk..kk+7, kk+16..kk+23}, hi lanes K {kk+8..kk+15, kk+24..kk+31}).
// D/C: v8f, lane holds 8 consecutive output channels m0+mi*16 + 8*(lane>=16).
template <int MT, int KD, bool RELU>
__device__ __forceinline__ void gemm_tiles(const __bf16* __restrict__ Wg,
                                           const float* __restrict__ bias,
                                           const __bf16* Bact, __bf16* Olds,
                                           int OD, int m0, int npix0, int lane) {
  const int l15   = lane & 15;
  const int lhalf = (lane >> 4) & 1;
  const int pix   = npix0 + l15;

  v8f acc[MT];
#pragma unroll
  for (int i = 0; i < MT; ++i)
#pragma unroll
    for (int j = 0; j < 8; ++j) acc[i][j] = 0.f;

#pragma unroll
  for (int kk = 0; kk < KD; kk += 32) {
    // B tile: two 16B LDS loads matching A's K layout (ds_load_b128 x2)
    const __bf16* bp = Bact + pix * KD + kk + lhalf * 8;
    v8bf blo = *(const v8bf*)(bp);
    v8bf bhi = *(const v8bf*)(bp + 16);
    v16bf b;
#pragma unroll
    for (int i = 0; i < 8; ++i) { b[i] = blo[i]; b[i + 8] = bhi[i]; }
#pragma unroll
    for (int mi = 0; mi < MT; ++mi) {
      const __bf16* wp = Wg + (m0 + mi * 16 + l15) * KD + kk + lhalf * 8;
      v8bf alo = *(const v8bf*)(wp);
      v8bf ahi = *(const v8bf*)(wp + 16);
      v16bf a;
#pragma unroll
      for (int i = 0; i < 8; ++i) { a[i] = alo[i]; a[i + 8] = ahi[i]; }
      acc[mi] = __builtin_amdgcn_wmma_f32_16x16x32_bf16(
          false, a, false, b, (short)0, acc[mi], false, false);
    }
  }

#pragma unroll
  for (int mi = 0; mi < MT; ++mi) {
    int ch = m0 + mi * 16 + lhalf * 8;
    v8f bv = *(const v8f*)(bias + ch);
    v8f r  = acc[mi] + bv;
    v8bf o;
#pragma unroll
    for (int i = 0; i < 8; ++i) {
      float t = r[i];
      if (RELU) t = t > 0.f ? t : 0.f;
      o[i] = (__bf16)t;
    }
    *(v8bf*)(Olds + pix * OD + ch) = o;  // 16B contiguous -> ds_store_b128
  }
}

__device__ __forceinline__ void layer1(const float* __restrict__ w1,
                                       const float* __restrict__ b1,
                                       __bf16* act1, int p, int obase,
                                       float p0, float p1, float p2) {
#pragma unroll
  for (int i = 0; i < 16; ++i) {
    int o = obase + i;
    float v = w1[o * 3 + 0] * p0 + w1[o * 3 + 1] * p1 + w1[o * 3 + 2] * p2 + b1[o];
    act1[p * 64 + o] = (__bf16)(v > 0.f ? v : 0.f);
  }
}

__global__ __launch_bounds__(256, 1)
void biup_fused(const float* __restrict__ x, const float* __restrict__ pose,
                const int* __restrict__ imY, const int* __restrict__ imX,
                const float* __restrict__ sigmar,
                const float* __restrict__ dw_w1, const float* __restrict__ dw_b1,
                const float* __restrict__ dw_b2, const float* __restrict__ dw_b3,
                const float* __restrict__ pw_w1, const float* __restrict__ pw_b1,
                const float* __restrict__ pw_b2, const float* __restrict__ pw_b3,
                const __bf16* __restrict__ wsbf, float* __restrict__ out) {
  extern __shared__ char smem[];
  __bf16* act1   = (__bf16*)(smem + ACT1_OFF);
  __bf16* act2   = (__bf16*)(smem + ACT2_OFF);
  __bf16* dwbuf  = (__bf16*)(smem + DWB_OFF);
  __bf16* pwbuf  = (__bf16*)(smem + PWB_OFF);
  float*  outbuf = (float*)(smem + OUTB_OFF);

  const int t      = threadIdx.x;
  const int lane   = t & 31;
  const int wave   = t >> 5;
  const int npix0  = (wave & 3) * 16;   // pixel tile for this wave
  const int msplit = wave >> 2;         // M split (0/1)
  const int p      = t & 63;            // block-local pixel for elementwise phases
  const int obase  = (t >> 6) * 16;
  const int gbase  = blockIdx.x * NPIX;
  const int gpix   = gbase + p;

  const float p0 = pose[gpix];
  const float p1 = pose[HOWO + gpix];
  const float p2 = pose[2 * HOWO + gpix];

  const __bf16* w2dw = wsbf + WS_DW2;
  const __bf16* w2pw = wsbf + WS_PW2;
  const __bf16* w3dw = wsbf + WS_DW3;
  const __bf16* w3pw = wsbf + WS_PW3;

  // ---------------- dw MLP: 3 -> 64 -> 256 -> 576 ----------------
  layer1(dw_w1, dw_b1, act1, p, obase, p0, p1, p2);
  __syncthreads();
  gemm_tiles<8, 64, true>(w2dw, dw_b2, act1, act2, 256, msplit * 128, npix0, lane);
  __syncthreads();
  gemm_tiles<18, 256, false>(w3dw, dw_b3, act2, dwbuf, 576, msplit * 288, npix0, lane);
  __syncthreads();

  // ---------------- pw MLP: 3 -> 64 -> 256 -> 192 ----------------
  layer1(pw_w1, pw_b1, act1, p, obase, p0, p1, p2);
  __syncthreads();
  gemm_tiles<8, 64, true>(w2pw, pw_b2, act1, act2, 256, msplit * 128, npix0, lane);
  __syncthreads();
  gemm_tiles<6, 256, false>(w3pw, pw_b3, act2, pwbuf, 192, msplit * 96, npix0, lane);
  __syncthreads();

  // ---------------- bilateral filter + 64->3 einsum ----------------
  if (t < 192) outbuf[t] = 0.f;
  __syncthreads();

  {
    const int cbase = (t >> 6) * 16;  // this thread owns 16 channels of pixel p
    const int iy = imY[gpix];
    const int ix = imX[gpix];
    int  off[9];
    bool val[9];
#pragma unroll
    for (int k = 0; k < 9; ++k) {
      int ry = iy + (k / 3) - 1;
      int rx = ix + (k % 3) - 1;
      val[k] = ((unsigned)ry < (unsigned)HI_IN) && ((unsigned)rx < (unsigned)WI_IN);
      off[k] = ry * WI_IN + rx;
    }

    float a0 = 0.f, a1 = 0.f, a2 = 0.f;
    const __bf16* dwp = dwbuf + p * 576;
    const __bf16* pwp = pwbuf + p * 192;

    for (int c = cbase; c < cbase + 16; ++c) {
      const float* xc = x + c * HWIN;
      float h[9], dwv[9];
#pragma unroll
      for (int k = 0; k < 9; ++k) h[k] = val[k] ? xc[off[k]] : 0.f;
#pragma unroll
      for (int k = 0; k < 9; ++k) dwv[k] = (float)dwp[c * 9 + k];

      // softmax(dw) -> weighted mean of taps
      float m1 = dwv[0];
#pragma unroll
      for (int k = 1; k < 9; ++k) m1 = fmaxf(m1, dwv[k]);
      float s1 = 0.f, num1 = 0.f;
#pragma unroll
      for (int k = 0; k < 9; ++k) {
        float e = __expf(dwv[k] - m1);
        s1 += e;
        num1 += e * h[k];
      }
      float mean = num1 / s1;

      float sg = sigmar[c];
      float bw[9];
      float m2 = -3.4e38f;
#pragma unroll
      for (int k = 0; k < 9; ++k) {
        float d = h[k] - mean;
        bw[k] = dwv[k] + sg * d * d;
        m2 = fmaxf(m2, bw[k]);
      }
      float s2 = 0.f, num2 = 0.f;
#pragma unroll
      for (int k = 0; k < 9; ++k) {
        float e = __expf(bw[k] - m2);
        s2 += e;
        num2 += e * h[k];
      }
      float xcv = num2 / s2;

      a0 += xcv * (float)pwp[c * 3 + 0];
      a1 += xcv * (float)pwp[c * 3 + 1];
      a2 += xcv * (float)pwp[c * 3 + 2];
    }
    atomicAdd(outbuf + p * 3 + 0, a0);  // ds_add_f32
    atomicAdd(outbuf + p * 3 + 1, a1);
    atomicAdd(outbuf + p * 3 + 2, a2);
  }
  __syncthreads();

  if (t < 192) {
    int o  = t >> 6;
    int pp = t & 63;
    out[o * HOWO + gbase + pp] = outbuf[pp * 3 + o];
  }
}

extern "C" void kernel_launch(void* const* d_in, const int* in_sizes, int n_in,
                              void* d_out, int out_size, void* d_ws, size_t ws_size,
                              hipStream_t stream) {
  const float* x      = (const float*)d_in[0];
  const float* pose   = (const float*)d_in[1];
  const int*   imY    = (const int*)d_in[2];
  const int*   imX    = (const int*)d_in[3];
  const float* sigmar = (const float*)d_in[4];
  const float* dw_w1  = (const float*)d_in[5];
  const float* dw_b1  = (const float*)d_in[6];
  const float* dw_w2  = (const float*)d_in[7];
  const float* dw_b2  = (const float*)d_in[8];
  const float* dw_w3  = (const float*)d_in[9];
  const float* dw_b3  = (const float*)d_in[10];
  const float* pw_w1  = (const float*)d_in[11];
  const float* pw_b1  = (const float*)d_in[12];
  const float* pw_w2  = (const float*)d_in[13];
  const float* pw_b2  = (const float*)d_in[14];
  const float* pw_w3  = (const float*)d_in[15];
  const float* pw_b3  = (const float*)d_in[16];
  float* out = (float*)d_out;
  __bf16* wsbf = (__bf16*)d_ws;

  biup_cvt_weights<<<(WS_ELEMS + 255) / 256, 256, 0, stream>>>(dw_w2, dw_w3, pw_w2,
                                                               pw_w3, wsbf);
  biup_fused<<<NBLKS, 256, SMEM_BYTES, stream>>>(
      x, pose, imY, imX, sigmar, dw_w1, dw_b1, dw_b2, dw_b3, pw_w1, pw_b1, pw_b2,
      pw_b3, wsbf, out);
}
